// MiniGPT4OAttention_46059229282609
// MI455X (gfx1250) — matrix operand order repaired and checked by reading, not compiled, not run on hardware
//
#include <hip/hip_runtime.h>
#include <hip/hip_bf16.h>

typedef __attribute__((ext_vector_type(16))) __bf16 v16bf;
typedef __attribute__((ext_vector_type(8)))  __bf16 v8bf;
typedef __attribute__((ext_vector_type(4)))  __bf16 v4bf;
typedef __attribute__((ext_vector_type(8)))  float  v8f;
typedef __attribute__((ext_vector_type(4)))  float  v4f;
typedef int v4i32 __attribute__((vector_size(16)));   // gcc-style, matches builtin

#define S_LEN  4096
#define HID    2048
#define NHEADS 8
#define NKV    2
#define HD     256
#define WINDOW 512

#if defined(__gfx1250__)
# if __has_builtin(__builtin_amdgcn_global_load_async_to_lds_b128) && \
     __has_builtin(__builtin_amdgcn_s_wait_asynccnt)
#  define HAVE_ASYNC_LDS 1
# endif
#endif
#ifndef HAVE_ASYNC_LDS
# define HAVE_ASYNC_LDS 0
#endif

#if HAVE_ASYNC_LDS
typedef __attribute__((address_space(1))) v4i32 gv4i;   // global (AS1)
typedef __attribute__((address_space(3))) v4i32 lv4i;   // LDS (AS3)
#endif

__device__ __forceinline__ v16bf cat8(v8bf lo, v8bf hi) {
  return __builtin_shufflevector(lo, hi, 0,1,2,3,4,5,6,7,8,9,10,11,12,13,14,15);
}
__device__ __forceinline__ v16bf zero16() {
  v16bf z;
  for (int e = 0; e < 16; ++e) z[e] = (__bf16)0.0f;
  return z;
}

// ---------------- elementwise f32 -> bf16 convert --------------------------
__global__ __launch_bounds__(256) void cvt_bf16(
    const float* __restrict__ in, __bf16* __restrict__ out, int n4) {
  int i = blockIdx.x * 256 + threadIdx.x;
  if (i >= n4) return;
  v4f v = *(const v4f*)(in + (size_t)i * 4);
  v4bf o;
  for (int e = 0; e < 4; ++e) o[e] = (__bf16)v[e];
  *(v4bf*)(out + (size_t)i * 4) = o;
}

// -------- f32 (K,N) -> bf16 (N,K) transpose-convert (LDS tiled) ------------
__global__ __launch_bounds__(256) void cvt_transpose_bf16(
    const float* __restrict__ in, __bf16* __restrict__ out, int K, int N) {
  __shared__ float t[32][33];
  const int k0 = blockIdx.y * 32, n0 = blockIdx.x * 32;
  const int tx = threadIdx.x & 31, ty = threadIdx.x >> 5;  // 32x8
  for (int i = 0; i < 4; ++i)
    t[ty + i * 8][tx] = in[(size_t)(k0 + ty + i * 8) * N + n0 + tx];
  __syncthreads();
  for (int i = 0; i < 4; ++i)
    out[(size_t)(n0 + ty + i * 8) * K + k0 + tx] = (__bf16)t[tx][ty + i * 8];
}

// -------- GEMM: C[M,N] = A[M,K] @ Bt[N,K]^T ; bf16 in, f32 out -------------
// Both operands K-major so every WMMA fragment is two contiguous 16B LDS reads,
// and tile staging is pure 16B chunks (async-to-LDS when available).
__global__ __launch_bounds__(128) void gemm_bf16(
    const __bf16* __restrict__ A, const __bf16* __restrict__ Bt,
    float* __restrict__ C, int M, int N, int K) {
  __shared__ __bf16 As[64][32];
  __shared__ __bf16 Bs[64][32];
  const int tid  = threadIdx.x;
  const int lane = tid & 31;
  const int wid  = tid >> 5;
  const int wm = wid >> 1, wn = wid & 1;
  const int half = lane >> 4, lr = lane & 15;
  const int bm0 = blockIdx.y * 64;
  const int bn0 = blockIdx.x * 64;

  v8f acc[2][2];
  for (int i = 0; i < 2; ++i)
    for (int j = 0; j < 2; ++j) acc[i][j] = {};

  const int nk = K / 32;
  for (int k0 = 0; k0 < nk; ++k0) {
    const int kb = k0 * 32;
    // 64 rows x 4 chunks of 16B per tile; 2 chunks per thread per tile
    for (int c = 0; c < 2; ++c) {
      const int chunk = tid + c * 128;         // 0..255
      const int r = chunk >> 2, cc = chunk & 3;
      const __bf16* ga = A  + (size_t)(bm0 + r) * K + kb + cc * 8;
      const __bf16* gb = Bt + (size_t)(bn0 + r) * K + kb + cc * 8;
#if HAVE_ASYNC_LDS
      __builtin_amdgcn_global_load_async_to_lds_b128(
          (gv4i*)ga, (lv4i*)&As[r][cc * 8], 0, 0);
      __builtin_amdgcn_global_load_async_to_lds_b128(
          (gv4i*)gb, (lv4i*)&Bs[r][cc * 8], 0, 0);
#else
      *(v8bf*)&As[r][cc * 8] = *(const v8bf*)ga;
      *(v8bf*)&Bs[r][cc * 8] = *(const v8bf*)gb;
#endif
    }
#if HAVE_ASYNC_LDS
    __builtin_amdgcn_s_wait_asynccnt(0);
#endif
    __syncthreads();

    v16bf af[2], bfv[2];
    for (int ms = 0; ms < 2; ++ms) {
      const int row = wm * 32 + ms * 16 + lr;
      v8bf lo = *(const v8bf*)&As[row][8 * half];       // K = 8h .. 8h+7
      v8bf hi = *(const v8bf*)&As[row][16 + 8 * half];  // K = 16+8h ..
      af[ms] = cat8(lo, hi);
    }
    for (int ns = 0; ns < 2; ++ns) {
      const int col = wn * 32 + ns * 16 + lr;
      v8bf lo = *(const v8bf*)&Bs[col][8 * half];       // K = 8h .. 8h+7
      v8bf hi = *(const v8bf*)&Bs[col][16 + 8 * half];
      bfv[ns] = cat8(lo, hi);
    }
    for (int ms = 0; ms < 2; ++ms)
      for (int ns = 0; ns < 2; ++ns)
        acc[ms][ns] = __builtin_amdgcn_wmma_f32_16x16x32_bf16(
            false, af[ms], false, bfv[ns], (short)0, acc[ms][ns], false, false);
    __syncthreads();
  }
  for (int ms = 0; ms < 2; ++ms)
    for (int ns = 0; ns < 2; ++ns)
      for (int r = 0; r < 8; ++r) {
        const int row = bm0 + wm * 32 + ms * 16 + r + 8 * half;
        const int col = bn0 + wn * 32 + ns * 16 + lr;
        C[(size_t)row * N + col] = acc[ms][ns][r];
      }
}

// -------- fused per-head RMS-norm (+RoPE for q/k, transpose for v) ---------
__global__ __launch_bounds__(256) void rmsnorm_rope(
    const float* __restrict__ X, const float* __restrict__ w,
    const float* __restrict__ cosb, const float* __restrict__ sinb,
    __bf16* __restrict__ out, int nheads, int vmode) {
  __shared__ float red[256];
  __shared__ float nrm[256];
  const int s = blockIdx.x, h = blockIdx.y, d = threadIdx.x;
  float x = X[(size_t)s * (nheads * HD) + h * HD + d];
  red[d] = x * x;
  __syncthreads();
  for (int off = 128; off > 0; off >>= 1) {
    if (d < off) red[d] += red[d + off];
    __syncthreads();
  }
  float scale = rsqrtf(red[0] * (1.0f / HD) + 1e-6f);
  if (vmode) {
    out[((size_t)h * HD + d) * S_LEN + s] = (__bf16)(x * scale);
  } else {
    float n = x * scale * w[d];
    nrm[d] = n;
    __syncthreads();
    float rot = (d < 128) ? -nrm[d + 128] : nrm[d - 128];
    float o = n * cosb[s * HD + d] + rot * sinb[s * HD + d];
    out[((size_t)h * S_LEN + s) * HD + d] = (__bf16)o;
  }
}

// -------- flash-style sliding-window attention, 1 wave / 16-query tile -----
__global__ __launch_bounds__(32) void attn_fwd(
    const __bf16* __restrict__ Qb,   // (NH, S, HD)
    const __bf16* __restrict__ Kb,   // (NKV, S, HD)
    const __bf16* __restrict__ Vt,   // (NKV, HD, S)  transposed
    __bf16* __restrict__ O) {        // (S, NH*HD) bf16: feeds out-proj GEMM
  __shared__ float ps[16][16];
  const int lane = threadIdx.x;
  const int half = lane >> 4, lr = lane & 15;
  const int qt = blockIdx.x, h = blockIdx.y;
  const int kvh = h >> 2;            // G = NH/NKV = 4
  const int qm0 = qt * 16;

  // Q fragments: 8 K-chunks of 32 covering D=256; all 16B vector loads
  v16bf qf[8];
  const __bf16* qrow = Qb + ((size_t)h * S_LEN + qm0 + lr) * HD;
  for (int c = 0; c < 8; ++c) {
    v8bf lo = *(const v8bf*)(qrow + c * 32 + 8 * half);
    v8bf hi = *(const v8bf*)(qrow + c * 32 + 16 + 8 * half);
    qf[c] = cat8(lo, hi);
  }

  v8f acc[16];
  for (int t = 0; t < 16; ++t) acc[t] = {};
  float m_i[8], l_i[8];
  for (int r = 0; r < 8; ++r) { m_i[r] = -3.0e38f; l_i[r] = 0.0f; }

  int kn_lo = qm0 - (WINDOW - 1);
  if (kn_lo < 0) kn_lo = 0;
  kn_lo &= ~15;
  for (int kn = kn_lo; kn <= qm0; kn += 16) {
    // scores 16x16 = Q @ K^T (fp32 accum; reference has no 1/sqrt(d))
    v8f sc = {};
    const __bf16* krow = Kb + ((size_t)kvh * S_LEN + kn + lr) * HD;
    for (int c = 0; c < 8; ++c) {
      v8bf lo = *(const v8bf*)(krow + c * 32 + 8 * half);
      v8bf hi = *(const v8bf*)(krow + c * 32 + 16 + 8 * half);
      sc = __builtin_amdgcn_wmma_f32_16x16x32_bf16(
          false, qf[c], false, cat8(lo, hi), (short)0, sc, false, false);
    }
    // online softmax; row M = r + 8*half lives across 16 lanes of a half
    const int j = kn + lr;
    for (int r = 0; r < 8; ++r) {
      const int i = qm0 + r + 8 * half;
      float sv = sc[r];
      bool ok = (j <= i) && (i - j < WINDOW);
      sv = ok ? sv : -3.0e38f;
      float rmax = sv;
      rmax = fmaxf(rmax, __shfl_xor(rmax, 1));
      rmax = fmaxf(rmax, __shfl_xor(rmax, 2));
      rmax = fmaxf(rmax, __shfl_xor(rmax, 4));
      rmax = fmaxf(rmax, __shfl_xor(rmax, 8));
      float mnew = fmaxf(m_i[r], rmax);
      float p = ok ? __expf(sv - mnew) : 0.0f;
      float rsum = p;
      rsum += __shfl_xor(rsum, 1);
      rsum += __shfl_xor(rsum, 2);
      rsum += __shfl_xor(rsum, 4);
      rsum += __shfl_xor(rsum, 8);
      float corr = __expf(m_i[r] - mnew);
      l_i[r] = l_i[r] * corr + rsum;
      m_i[r] = mnew;
      for (int t = 0; t < 16; ++t) acc[t][r] *= corr;
      ps[r + 8 * half][lr] = p;
    }
    __syncthreads();
    // P as A-fragment: valid keys live in elements 0..7 (K = 8*half + e)
    v16bf pa = zero16();
    {
      const float* prow = &ps[lr][8 * half];
      for (int e = 0; e < 8; ++e) pa[e] = (__bf16)prow[e];
    }
    __syncthreads();
    // acc[t] += P @ V ; Vt rows contiguous along keys; upper K half zero
    for (int t = 0; t < 16; ++t) {
      const __bf16* vrow = Vt + ((size_t)kvh * HD + t * 16 + lr) * S_LEN + kn;
      v8bf v0 = *(const v8bf*)(vrow + 8 * half);
      v8bf v1 = *(const v8bf*)(vrow + 8);
      v16bf vf = cat8(v0, v1);
      acc[t] = __builtin_amdgcn_wmma_f32_16x16x32_bf16(
          false, pa, false, vf, (short)0, acc[t], false, false);
    }
  }
  for (int t = 0; t < 16; ++t)
    for (int r = 0; r < 8; ++r) {
      const int row = qm0 + r + 8 * half;
      const int col = h * HD + t * 16 + lr;
      O[(size_t)row * HID + col] = (__bf16)(acc[t][r] / l_i[r]);
    }
}

extern "C" void kernel_launch(void* const* d_in, const int* in_sizes, int n_in,
                              void* d_out, int out_size, void* d_ws, size_t ws_size,
                              hipStream_t stream) {
  (void)in_sizes; (void)n_in; (void)out_size; (void)ws_size;
  const float* hs   = (const float*)d_in[0];
  const float* w_q  = (const float*)d_in[1];
  const float* w_k  = (const float*)d_in[2];
  const float* w_v  = (const float*)d_in[3];
  const float* w_o  = (const float*)d_in[4];
  const float* qnw  = (const float*)d_in[5];
  const float* knw  = (const float*)d_in[6];
  const float* cosb = (const float*)d_in[7];
  const float* sinb = (const float*)d_in[8];
  float* out = (float*)d_out;

  char* ws = (char*)d_ws;
  const size_t MB = (size_t)1 << 20;
  __bf16* hsb = (__bf16*)(ws);             // 16 MB  (S, H) bf16
  __bf16* wqT = (__bf16*)(ws + 16 * MB);   //  8 MB  (N=2048, K=2048)
  __bf16* wkT = (__bf16*)(ws + 24 * MB);   //  2 MB  (N=512,  K=2048)
  __bf16* wvT = (__bf16*)(ws + 26 * MB);   //  2 MB
  __bf16* woT = (__bf16*)(ws + 28 * MB);   //  8 MB
  float*  Qf  = (float*) (ws + 36 * MB);   // 32 MB  (S, 2048) f32
  __bf16* abf = (__bf16*)(ws + 36 * MB);   // 16 MB  aliases Qf (dead by then)
  float*  Kf  = (float*) (ws + 68 * MB);   //  8 MB
  float*  Vf  = (float*) (ws + 76 * MB);   //  8 MB
  __bf16* Qb  = (__bf16*)(ws + 84 * MB);   // 16 MB  (NH, S, HD)
  __bf16* Kb  = (__bf16*)(ws + 100 * MB);  //  4 MB  (NKV, S, HD)
  __bf16* Vt  = (__bf16*)(ws + 104 * MB);  //  4 MB  (NKV, HD, S)

  // 1) one-shot bf16 conversions (weights transposed to K-major)
  cvt_bf16<<<(S_LEN * HID / 4 + 255) / 256, 256, 0, stream>>>(hs, hsb, S_LEN * HID / 4);
  cvt_transpose_bf16<<<dim3(HID / 32, HID / 32), 256, 0, stream>>>(w_q, wqT, HID, HID);
  cvt_transpose_bf16<<<dim3((NKV * HD) / 32, HID / 32), 256, 0, stream>>>(w_k, wkT, HID, NKV * HD);
  cvt_transpose_bf16<<<dim3((NKV * HD) / 32, HID / 32), 256, 0, stream>>>(w_v, wvT, HID, NKV * HD);
  cvt_transpose_bf16<<<dim3(HID / 32, HID / 32), 256, 0, stream>>>(w_o, woT, HID, HID);

  // 2) QKV projections
  gemm_bf16<<<dim3(HID / 64, S_LEN / 64), 128, 0, stream>>>(hsb, wqT, Qf, S_LEN, HID, HID);
  gemm_bf16<<<dim3((NKV * HD) / 64, S_LEN / 64), 128, 0, stream>>>(hsb, wkT, Kf, S_LEN, NKV * HD, HID);
  gemm_bf16<<<dim3((NKV * HD) / 64, S_LEN / 64), 128, 0, stream>>>(hsb, wvT, Vf, S_LEN, NKV * HD, HID);

  // 3) per-head RMS-norm (+RoPE), layouts tuned for WMMA fragment loads
  rmsnorm_rope<<<dim3(S_LEN, NHEADS), 256, 0, stream>>>(Qf, qnw, cosb, sinb, Qb, NHEADS, 0);
  rmsnorm_rope<<<dim3(S_LEN, NKV),    256, 0, stream>>>(Kf, knw, cosb, sinb, Kb, NKV, 0);
  rmsnorm_rope<<<dim3(S_LEN, NKV),    256, 0, stream>>>(Vf, nullptr, nullptr, nullptr, Vt, NKV, 1);

  // 4) sliding-window attention -> bf16 activation for out-proj
  attn_fwd<<<dim3(S_LEN / 16, NHEADS), 32, 0, stream>>>(Qb, Kb, Vt, abf);

  // 5) output projection
  gemm_bf16<<<dim3(HID / 64, S_LEN / 64), 128, 0, stream>>>(abf, woT, out, S_LEN, HID, HID);
}
